// LagAwarePeerBlock_8572754723134
// MI455X (gfx1250) — matrix operand order repaired and checked by reading, not compiled, not run on hardware
//
#include <hip/hip_runtime.h>
#include <hip/hip_fp16.h>

typedef __attribute__((ext_vector_type(16))) _Float16 v16h;
typedef __attribute__((ext_vector_type(8)))  float    v8f;

#define HDIM 256
#define B_   4
#define N_   64
#define T_   384
#define L_   3
#define NU_  500
#define MT_  (B_ * T_)   // 1536 rows for all small GEMMs

// ---------------------------------------------------------------------------
// WMMA GEMM: C[M,256] = act( X[M,256] @ W^T + bias * scale ) + residual
// W is [256,256] row-major; row n of W supplies column n of B (contiguous).
// One wave computes a 16x64 output strip: A fragment loaded once per k-chunk,
// reused for 4 WMMAs (4 f32 accumulators).
// ---------------------------------------------------------------------------
__global__ __launch_bounds__(128) void gemm_wmma_kernel(
    const float* __restrict__ X, const float* __restrict__ W,
    const float* __restrict__ bias, const float* __restrict__ bias_scale,
    const float* __restrict__ residual, float* __restrict__ C,
    int M, int act)
{
    const int wave = threadIdx.x >> 5;
    const int lane = threadIdx.x & 31;
    const int nWT = (M / 16) * (HDIM / 64);
    const int wt = blockIdx.x * 4 + wave;
    if (wt >= nWT) return;                  // wave-uniform: EXEC stays all-1s

    const int rowBase = (wt / (HDIM / 64)) * 16;
    const int colBase = (wt % (HDIM / 64)) * 64;

    const bool hi = lane >= 16;
    const int  mn = lane & 15;              // A row / B col within tile

    const float* arow = X + (size_t)(rowBase + mn) * HDIM;
    v8f acc[4] = {v8f{}, v8f{}, v8f{}, v8f{}};

    for (int k0 = 0; k0 < HDIM; k0 += 32) {
        // A fragment 16x32 f16: lane<16 K=0..15; lane>=16 K=16..31
        v16h a;
        const int ka = k0 + (hi ? 16 : 0);
        #pragma unroll
        for (int i = 0; i < 8; ++i) a[i]     = (_Float16)arow[ka + i];
        #pragma unroll
        for (int i = 0; i < 8; ++i) a[8 + i] = (_Float16)arow[ka + 8 + i];

        // B fragments 32x16 f16: lanes 0-15 K=k0..k0+15, lanes 16-31 K=k0+16..31
        const int kb = k0 + (hi ? 16 : 0);
        const float* bbase = W + (size_t)(colBase + mn) * HDIM + kb;
        #pragma unroll
        for (int j = 0; j < 4; ++j) {
            const float* brow = bbase + (size_t)j * 16 * HDIM;
            v16h bf;
            #pragma unroll
            for (int i = 0; i < 16; ++i) bf[i] = (_Float16)brow[i];
            acc[j] = __builtin_amdgcn_wmma_f32_16x16x32_f16(
                false, a, false, bf, (short)0, acc[j], false, false);
        }
    }

    // C/D layout: VGPR v -> M = v + (hi?8:0), N = lane&15
    const int mOut = rowBase + (hi ? 8 : 0);
    #pragma unroll
    for (int j = 0; j < 4; ++j) {
        const int col = colBase + j * 16 + mn;
        const float bc = bias ? bias[col] : 0.0f;
        #pragma unroll
        for (int v = 0; v < 8; ++v) {
            const int row = mOut + v;
            float val = acc[j][v];
            if (bias) val += bc * (bias_scale ? bias_scale[row] : 1.0f);
            if (act == 1) val = (val > 0.0f) ? val : (expf(val) - 1.0f);   // ELU
            if (residual) val += residual[(size_t)row * HDIM + col];
            C[(size_t)row * HDIM + col] = val;
        }
    }
}

// ---------------------------------------------------------------------------
// 256x256 transpose (16x16 LDS tiles) so the Qt GEMM can use the contiguous
// C = X @ W^T layout.
// ---------------------------------------------------------------------------
__global__ __launch_bounds__(256) void transpose_kernel(
    const float* __restrict__ in, float* __restrict__ out)
{
    __shared__ float tile[16][17];
    const int bx = blockIdx.x & 15;
    const int by = blockIdx.x >> 4;
    const int tx = threadIdx.x & 15;
    const int ty = threadIdx.x >> 4;
    tile[ty][tx] = in[(size_t)(by * 16 + ty) * HDIM + bx * 16 + tx];
    __syncthreads();
    out[(size_t)(bx * 16 + ty) * HDIM + by * 16 + tx] = tile[tx][ty];
}

// ---------------------------------------------------------------------------
// Logits + masks + top-8 + softmax + weighted delta accumulation.
// One 256-thread block per (b,t).
// ---------------------------------------------------------------------------
__global__ __launch_bounds__(256) void attn_topk_kernel(
    const float* __restrict__ Qt, const float* __restrict__ peer_h,
    const float* __restrict__ alpha_p, const float* __restrict__ lrs,
    const unsigned char* __restrict__ peer_mask,
    const unsigned char* __restrict__ rmask,
    const int* __restrict__ target_id, const int* __restrict__ peer_id,
    float* __restrict__ dbuf, float* __restrict__ wsum_buf)
{
    const int bt = blockIdx.x;
    const int b  = bt / T_;
    const int t  = bt % T_;
    const int tid  = threadIdx.x;
    const int lane = tid & 31;
    const int wv   = tid >> 5;

    __shared__ __align__(16) float qs[HDIM];
    __shared__ float slog[N_ * L_];      // 192 logits
    __shared__ float s_w[8];
    __shared__ int   s_i[8];
    __shared__ float s_wsum;

    const int lags[3] = {1, 5, 21};
    const float alpha = alpha_p[0];
    const int tgt = target_id[b];
    const float NEG_INF = -__builtin_inff();

    // Qt row scaled by 1/sqrt(H)=1/16  (wk_b term cancels: uniform logit shift)
    qs[tid] = Qt[(size_t)bt * HDIM + tid] * 0.0625f;
    __syncthreads();

    const float4 q0 = ((const float4*)qs)[lane * 2];
    const float4 q1 = ((const float4*)qs)[lane * 2 + 1];

    // ---- logits: each wave handles 24 (n,l) pairs, full-wave dot each ----
    for (int p = 0; p < 24; ++p) {
        const int nl = wv * 24 + p;
        const int n = nl / 3, l = nl % 3;
        const int lag = lags[l];

        // prefetch next pair's peer row (wave-uniform guard) -> global_prefetch_b8
        if (p + 1 < 24) {
            const int nl2 = nl + 1;
            const int n2 = nl2 / 3, l2 = nl2 % 3;
            const int lag2 = lags[l2];
            if (t >= lag2) {
                const float* kr2 =
                    peer_h + (((size_t)(b * N_ + n2)) * T_ + (t - lag2)) * HDIM;
                __builtin_prefetch(kr2 + lane * 8, 0, 1);
            }
        }

        const int pid = peer_id[b * N_ + n];
        const bool valid = (t >= lag) && (peer_mask[b * N_ + n] != 0) &&
                           (rmask[((size_t)l * NU_ + pid) * NU_ + tgt] != 0);
        float out;
        if (!valid) {
            out = NEG_INF;
        } else {
            const float* kr =
                peer_h + (((size_t)(b * N_ + n)) * T_ + (t - lag)) * HDIM;
            const float4 a0 = ((const float4*)kr)[lane * 2];
            const float4 a1 = ((const float4*)kr)[lane * 2 + 1];
            float d = a0.x * q0.x + a0.y * q0.y + a0.z * q0.z + a0.w * q0.w
                    + a1.x * q1.x + a1.y * q1.y + a1.z * q1.z + a1.w * q1.w;
            for (int off = 16; off > 0; off >>= 1) d += __shfl_down(d, off, 32);
            out = d + alpha * lrs[((size_t)l * NU_ + pid) * NU_ + tgt];
        }
        if (lane == 0) slog[nl] = out;
    }
    __syncthreads();

    // ---- wave 0: iterative top-8 (low-index tie-break, like lax.top_k) ----
    if (tid < 32) {
        float vals[6];
        const int base = lane * 6;
        #pragma unroll
        for (int j = 0; j < 6; ++j) vals[j] = slog[base + j];

        for (int k = 0; k < 8; ++k) {
            float bv = vals[0]; int bi = base;
            #pragma unroll
            for (int j = 1; j < 6; ++j)
                if (vals[j] > bv) { bv = vals[j]; bi = base + j; }
            for (int off = 16; off > 0; off >>= 1) {
                const float ov = __shfl_down(bv, off, 32);
                const int   oi = __shfl_down(bi, off, 32);
                if (ov > bv || (ov == bv && oi < bi)) { bv = ov; bi = oi; }
            }
            bv = __shfl(bv, 0, 32);
            bi = __shfl(bi, 0, 32);
            if (bi >= base && bi < base + 6) vals[bi - base] = NEG_INF;
            if (lane == 0) { s_w[k] = bv; s_i[k] = bi; }
        }
        if (lane == 0) {
            bool allinf = true;
            float mx = -3.0e38f;
            float safe[8];
            for (int k = 0; k < 8; ++k) {
                float v = s_w[k];
                if (v == NEG_INF) v = -1e9f; else allinf = false;
                safe[k] = v;
                if (v > mx) mx = v;
            }
            float s = 0.0f, w[8];
            for (int k = 0; k < 8; ++k) { w[k] = expf(safe[k] - mx); s += w[k]; }
            const float inv = allinf ? 0.0f : 1.0f / s;
            for (int k = 0; k < 8; ++k) s_w[k] = w[k] * inv;
            s_wsum = allinf ? 0.0f : 1.0f;
        }
    }
    __syncthreads();

    // ---- weighted delta vector: d = sum_k w_k * (x_now - x_lag) ----
    float acc = 0.0f;
    for (int k = 0; k < 8; ++k) {
        const float w = s_w[k];            // uniform across block
        if (w != 0.0f) {
            const int nl = s_i[k];
            const int n = nl / 3, l = nl % 3;
            int tc = t - lags[l]; if (tc < 0) tc = 0;
            const float* prow = peer_h + ((size_t)(b * N_ + n) * T_) * HDIM;
            acc += w * (prow[(size_t)t * HDIM + tid] - prow[(size_t)tc * HDIM + tid]);
        }
    }
    dbuf[(size_t)bt * HDIM + tid] = acc;
    if (tid == 0) wsum_buf[bt] = s_wsum;
}

// ---------------------------------------------------------------------------
// LayerNorm over H=256, one block per row.
// ---------------------------------------------------------------------------
__global__ __launch_bounds__(256) void layernorm_kernel(
    const float* __restrict__ x, const float* __restrict__ g,
    const float* __restrict__ bta, float* __restrict__ out)
{
    const int row = blockIdx.x;
    const int h = threadIdx.x;
    const float v = x[(size_t)row * HDIM + h];
    __shared__ float rs[HDIM];
    __shared__ float rq[HDIM];
    rs[h] = v;
    rq[h] = v * v;
    __syncthreads();
    for (int off = 128; off > 0; off >>= 1) {
        if (h < off) { rs[h] += rs[h + off]; rq[h] += rq[h + off]; }
        __syncthreads();
    }
    const float mu  = rs[0] * (1.0f / HDIM);
    const float var = rq[0] * (1.0f / HDIM) - mu * mu;
    const float r   = rsqrtf(var + 1e-5f);
    out[(size_t)row * HDIM + h] = (v - mu) * r * g[h] + bta[h];
}

// ---------------------------------------------------------------------------
extern "C" void kernel_launch(void* const* d_in, const int* in_sizes, int n_in,
                              void* d_out, int out_size, void* d_ws, size_t ws_size,
                              hipStream_t stream) {
    (void)in_sizes; (void)n_in; (void)out_size; (void)ws_size;
    const float* target_h = (const float*)d_in[0];
    const float* peer_h   = (const float*)d_in[1];
    const float* wq_w = (const float*)d_in[2];
    const float* wq_b = (const float*)d_in[3];
    const float* wk_w = (const float*)d_in[4];
    // wk_b (d_in[5]) adds a uniform shift to all logits at fixed (b,t):
    // invariant under top-k and softmax -> provably cancels, not needed.
    const float* wv_w = (const float*)d_in[6];
    const float* wv_b = (const float*)d_in[7];
    const float* f1_w = (const float*)d_in[8];
    const float* f1_b = (const float*)d_in[9];
    const float* f2_w = (const float*)d_in[10];
    const float* f2_b = (const float*)d_in[11];
    const float* ln_g = (const float*)d_in[12];
    const float* ln_b = (const float*)d_in[13];
    const float* alpha = (const float*)d_in[14];
    const float* lrs   = (const float*)d_in[15];
    const unsigned char* peer_mask = (const unsigned char*)d_in[16];
    const unsigned char* rmask     = (const unsigned char*)d_in[17];
    const int* tgt_id = (const int*)d_in[18];
    const int* pid    = (const int*)d_in[19];

    float* ws  = (float*)d_ws;
    float* Qm   = ws;                    // [1536,256]
    float* Qt   = Qm  + (size_t)MT_ * HDIM;
    float* db   = Qt  + (size_t)MT_ * HDIM;
    float* lly  = db  + (size_t)MT_ * HDIM;
    float* hid  = lly + (size_t)MT_ * HDIM;
    float* xb   = hid + (size_t)MT_ * HDIM;
    float* wkT  = xb  + (size_t)MT_ * HDIM;   // [256,256]
    float* wsum = wkT + (size_t)HDIM * HDIM;  // [1536]

    const int gemmGrid = ((MT_ / 16) * (HDIM / 64) + 3) / 4;  // 4 waves/block

    // wkT = wk_w^T  (so Qt GEMM can use the contiguous X @ W^T path)
    transpose_kernel<<<256, 256, 0, stream>>>(wk_w, wkT);
    // Q = target_h @ wq_w^T + wq_b
    gemm_wmma_kernel<<<gemmGrid, 128, 0, stream>>>(target_h, wq_w, wq_b, nullptr, nullptr, Qm, MT_, 0);
    // Qt = Q @ wk_w = Q @ wkT^T   (Wk hoisted onto Q)
    gemm_wmma_kernel<<<gemmGrid, 128, 0, stream>>>(Qm, wkT, nullptr, nullptr, nullptr, Qt, MT_, 0);
    // logits + top-8 + softmax + weighted delta
    attn_topk_kernel<<<MT_, 256, 0, stream>>>(Qt, peer_h, alpha, lrs, peer_mask, rmask, tgt_id, pid, db, wsum);
    // ll_y = d @ wv_w^T + wsum * wv_b
    gemm_wmma_kernel<<<gemmGrid, 128, 0, stream>>>(db, wv_w, wv_b, wsum, nullptr, lly, MT_, 0);
    // hid = elu(ll_y @ f1_w^T + f1_b)
    gemm_wmma_kernel<<<gemmGrid, 128, 0, stream>>>(lly, f1_w, f1_b, nullptr, nullptr, hid, MT_, 1);
    // x = ll_y + hid @ f2_w^T + f2_b
    gemm_wmma_kernel<<<gemmGrid, 128, 0, stream>>>(hid, f2_w, f2_b, nullptr, lly, xb, MT_, 0);
    // out = LayerNorm(x) * g + b
    layernorm_kernel<<<MT_, 256, 0, stream>>>(xb, ln_g, ln_b, (float*)d_out);
}
